// MultiHeadAttention_25993142075914
// MI455X (gfx1250) — matrix operand order repaired and checked by reading, not compiled, not run on hardware
//
#include <hip/hip_runtime.h>

typedef __bf16 bf16_t;
typedef __attribute__((ext_vector_type(16))) __bf16 v16bf;
typedef __attribute__((ext_vector_type(8)))  __bf16 v8bf;
typedef __attribute__((ext_vector_type(8)))  float  v8f;
typedef __attribute__((ext_vector_type(4)))  int    v4i;

#define N_HEAD  16
#define D_HEAD  64
#define D_MODEL 1024
#define N_TOK   2048

union Frag16 { v16bf v; v8bf h[2]; };

// ---------- async global->LDS path (gfx1250 GLOBAL_LOAD_ASYNC_TO_LDS_B128) ----------
#if defined(__has_builtin)
#if __has_builtin(__builtin_amdgcn_global_load_async_to_lds_b128)
#define HAVE_ASYNC_LDS 1
#endif
#endif

#ifdef HAVE_ASYNC_LDS
typedef __attribute__((address_space(1))) v4i as1_v4i;   // global int4
typedef __attribute__((address_space(3))) v4i as3_v4i;   // LDS int4
__device__ __forceinline__ void async_copy_b128(const void* gsrc, void* ldst) {
  __builtin_amdgcn_global_load_async_to_lds_b128(
      (as1_v4i*)(unsigned long long)gsrc,
      (as3_v4i*)(unsigned int)(unsigned long long)ldst,
      0, 0);
}
__device__ __forceinline__ void async_wait0() {
#if __has_builtin(__builtin_amdgcn_s_wait_asynccnt)
  __builtin_amdgcn_s_wait_asynccnt(0);
#else
  asm volatile("s_wait_asynccnt 0x0" ::: "memory");
#endif
}
#endif

__device__ __forceinline__ void cvt4(const float* src, bf16_t* dst) {
  const float4 f = *(const float4*)src;
  dst[0] = (bf16_t)f.x; dst[1] = (bf16_t)f.y;
  dst[2] = (bf16_t)f.z; dst[3] = (bf16_t)f.w;
}
__device__ __forceinline__ void cvt4(const bf16_t* src, bf16_t* dst) {
  *(uint2*)dst = *(const uint2*)src;
}

// C[M,N] = scale * A[M,K] * Bt[N,K] + bias[N]
// A: fp32 (global loads -> regs -> cvt -> LDS, overlapped with WMMA) or bf16 (async DMA).
// Bt: bf16, K-contiguous, async DMA straight to LDS.
// Double-buffered LDS software pipeline; batched over blockIdx.z via element strides.
template<int BM, int BN, int WM, int WN, typename AT, typename OT>
__global__ __launch_bounds__(256) void gemm_bf16_wmma(
    const AT* __restrict__ A, long long aBatch, int lda,
    const bf16_t* __restrict__ B, long long bBatch, int ldb,
    OT* __restrict__ C, long long cBatch, int ldc,
    const float* __restrict__ bias,
    int K, float scale)
{
  constexpr int BK = 32;
  constexpr int WAVES_N = BN / WN;
  constexpr int MT = WM / 16;
  constexpr int NT = WN / 16;
  constexpr int AIT4 = (BM * BK) / (256 * 4);   // fp32-A staging chunks per tile
  constexpr int AIT8 = (BM * BK) / (256 * 8);   // bf16-A async chunks per tile
  constexpr int BIT8 = (BN * BK) / (256 * 8);   // bf16-B async chunks per tile
  constexpr int BIT4 = (BN * BK) / (256 * 4);
  __shared__ bf16_t lA[2][BM * BK];
  __shared__ bf16_t lB[2][BN * BK];

  A += (long long)blockIdx.z * aBatch;
  B += (long long)blockIdx.z * bBatch;
  C += (long long)blockIdx.z * cBatch;

  const int tid  = threadIdx.x;
  const int wave = tid >> 5;
  const int lane = tid & 31;
  const int hlf  = lane >> 4;   // lane half (0: lanes 0-15, 1: lanes 16-31)
  const int l16  = lane & 15;
  const int wm   = wave / WAVES_N;
  const int wn   = wave % WAVES_N;
  const int blockM = blockIdx.x * BM;
  const int blockN = blockIdx.y * BN;

  v8f acc[MT][NT];
  const v8f vzero = {0.f, 0.f, 0.f, 0.f, 0.f, 0.f, 0.f, 0.f};
  for (int i = 0; i < MT; ++i)
    for (int j = 0; j < NT; ++j) acc[i][j] = vzero;

  float4 av[AIT4 > 0 ? AIT4 : 1];   // fp32-A in-flight registers

  // Phase 1: issue all global traffic for tile at k-offset kk (no waits).
  auto stage_load = [&](int kk, int buf) {
#ifdef HAVE_ASYNC_LDS
    if constexpr (sizeof(AT) == 2) {
#pragma unroll
      for (int it = 0; it < AIT8; ++it) {
        const int i = tid * 8 + it * 2048;
        const int r = i / BK, c = i % BK;
        async_copy_b128(A + (long long)(blockM + r) * lda + (kk + c), &lA[buf][i]);
      }
    } else {
#pragma unroll
      for (int it = 0; it < AIT4; ++it) {
        const int i = tid * 4 + it * 1024;
        const int r = i / BK, c = i % BK;
        av[it] = *(const float4*)(A + (long long)(blockM + r) * lda + (kk + c));
      }
      if (kk + BK < K)  // prefetch the tile after the one being staged
        __builtin_prefetch(
            A + (long long)(blockM + (tid >> 1)) * lda + (kk + BK + ((tid & 1) << 4)), 0, 3);
    }
#pragma unroll
    for (int it = 0; it < BIT8; ++it) {
      const int i = tid * 8 + it * 2048;
      const int r = i / BK, c = i % BK;
      async_copy_b128(B + (long long)(blockN + r) * ldb + (kk + c), &lB[buf][i]);
    }
#else
    // Fallback: synchronous convert/copy staging
    {
#pragma unroll
      for (int it = 0; it < AIT4; ++it) {
        const int i = tid * 4 + it * 1024;
        const int r = i / BK, c = i % BK;
        union { bf16_t b[4]; uint2 u; } t;
        cvt4(A + (long long)(blockM + r) * lda + (kk + c), t.b);
        *(uint2*)&lA[buf][i] = t.u;
      }
#pragma unroll
      for (int it = 0; it < BIT4; ++it) {
        const int i = tid * 4 + it * 1024;
        const int r = i / BK, c = i % BK;
        union { bf16_t b[4]; uint2 u; } t;
        cvt4(B + (long long)(blockN + r) * ldb + (kk + c), t.b);
        *(uint2*)&lB[buf][i] = t.u;
      }
    }
#endif
  };

  // Phase 2: land staged fp32-A registers into LDS as bf16 (after the WMMAs).
  auto stage_store = [&](int buf) {
#ifdef HAVE_ASYNC_LDS
    if constexpr (sizeof(AT) == 4) {
#pragma unroll
      for (int it = 0; it < AIT4; ++it) {
        const int i = tid * 4 + it * 1024;
        union { bf16_t b[4]; uint2 u; } t;
        t.b[0] = (bf16_t)av[it].x; t.b[1] = (bf16_t)av[it].y;
        t.b[2] = (bf16_t)av[it].z; t.b[3] = (bf16_t)av[it].w;
        *(uint2*)&lA[buf][i] = t.u;
      }
    }
#endif
  };

  // Prologue: stage tile 0 into buffer 0.
  stage_load(0, 0);
  stage_store(0);
#ifdef HAVE_ASYNC_LDS
  async_wait0();
#endif
  __syncthreads();

  const int nk = K / BK;
  for (int s = 0; s < nk; ++s) {
    const int buf = s & 1;
    const bool haveNext = (s + 1) < nk;
    if (haveNext) stage_load((s + 1) * BK, buf ^ 1);

    // ---- Fragments per ISA 7.12.2 (16-bit A 16x32; B with the same k order as A) ----
    v16bf afrag[MT], bfrag[NT];
#pragma unroll
    for (int mt = 0; mt < MT; ++mt) {
      const int row = wm * WM + mt * 16 + l16;
      Frag16 f;
      f.h[0] = *(const v8bf*)&lA[buf][row * BK + 8 * hlf];       // K = 8h .. 8h+7
      f.h[1] = *(const v8bf*)&lA[buf][row * BK + 16 + 8 * hlf];  // K = 16+8h .. 16+8h+7
      afrag[mt] = f.v;
    }
#pragma unroll
    for (int nt = 0; nt < NT; ++nt) {
      const int row = wn * WN + nt * 16 + l16;
      Frag16 f;
      f.h[0] = *(const v8bf*)&lB[buf][row * BK + 8 * hlf];       // K = 8h .. 8h+7
      f.h[1] = *(const v8bf*)&lB[buf][row * BK + 16 + 8 * hlf];  // K = 16+8h .. 16+8h+7
      bfrag[nt] = f.v;
    }
#pragma unroll
    for (int mt = 0; mt < MT; ++mt)
#pragma unroll
      for (int nt = 0; nt < NT; ++nt)
        acc[mt][nt] = __builtin_amdgcn_wmma_f32_16x16x32_bf16(
            false, afrag[mt], false, bfrag[nt], (short)0, acc[mt][nt],
            false, false);

    if (haveNext) stage_store(buf ^ 1);
#ifdef HAVE_ASYNC_LDS
    async_wait0();   // own async writes to buf^1 complete before the barrier
#endif
    __syncthreads();
  }

  // Epilogue: C/D layout — VGPR r: lanes 0-15 -> M=r, lanes 16-31 -> M=8+r; N = lane%16
#pragma unroll
  for (int mt = 0; mt < MT; ++mt) {
#pragma unroll
    for (int nt = 0; nt < NT; ++nt) {
      const int col = blockN + wn * WN + nt * 16 + l16;
      const float bb = bias ? bias[col] : 0.f;
#pragma unroll
      for (int r = 0; r < 8; ++r) {
        const int row = blockM + wm * WM + mt * 16 + 8 * hlf + r;
        C[(long long)row * ldc + col] = (OT)(acc[mt][nt][r] * scale + bb);
      }
    }
  }
}

// out[c*R + r] = (bf16) in[r*C + c]   (tiled transpose + convert)
template<typename T>
__global__ __launch_bounds__(256) void transpose_to_bf16(
    const T* __restrict__ in, bf16_t* __restrict__ out, int R, int C)
{
  __shared__ bf16_t tile[32][33];
  const int cb = blockIdx.x * 32;
  const int rb = blockIdx.y * 32;
  const int tx = threadIdx.x, ty = threadIdx.y;
  for (int i = ty; i < 32; i += 8)
    tile[i][tx] = (bf16_t)in[(long long)(rb + i) * C + cb + tx];
  __syncthreads();
  for (int i = ty; i < 32; i += 8)
    out[(long long)(cb + i) * R + rb + tx] = tile[tx][i];
}

// Column-wise (over q) online max/sum for u[h, q, k]. block (32,8); grid (N_TOK/32, N_HEAD)
__global__ __launch_bounds__(256) void softmax_stats(
    const float* __restrict__ u, float* __restrict__ mx, float* __restrict__ sm)
{
  const int col = blockIdx.x * 32 + threadIdx.x;
  const int h   = blockIdx.y;
  const float* base = u + (long long)h * N_TOK * N_TOK + col;
  float m = -__builtin_inff(), s = 0.f;
  for (int q = threadIdx.y; q < N_TOK; q += 8) {
    const float x  = base[(long long)q * N_TOK];
    const float nm = fmaxf(m, x);
    s = s * __expf(m - nm) + __expf(x - nm);
    m = nm;
  }
  __shared__ float sM[8][32];
  __shared__ float sS[8][32];
  sM[threadIdx.y][threadIdx.x] = m;
  sS[threadIdx.y][threadIdx.x] = s;
  __syncthreads();
  if (threadIdx.y == 0) {
    for (int i = 1; i < 8; ++i) {
      const float m2 = sM[i][threadIdx.x], s2 = sS[i][threadIdx.x];
      const float nm = fmaxf(m, m2);
      s = s * __expf(m - nm) + s2 * __expf(m2 - nm);
      m = nm;
    }
    mx[h * N_TOK + col] = m;
    sm[h * N_TOK + col] = s;
  }
}

// In-place: u = exp(u - mx[col]) / sm[col], float4-vectorized along k
__global__ __launch_bounds__(256) void softmax_apply(
    float* __restrict__ u, const float* __restrict__ mx, const float* __restrict__ sm)
{
  const long long base = ((long long)blockIdx.x * 256 + threadIdx.x) * 4;
  const int k = (int)(base & (N_TOK - 1));
  const int h = (int)(base >> 22);            // 2048*2048 = 2^22
  float4 x = *(float4*)&u[base];
  const float4 m = *(const float4*)&mx[h * N_TOK + k];
  const float4 s = *(const float4*)&sm[h * N_TOK + k];
  x.x = __expf(x.x - m.x) / s.x;
  x.y = __expf(x.y - m.y) / s.y;
  x.z = __expf(x.z - m.z) / s.z;
  x.w = __expf(x.w - m.w) / s.w;
  *(float4*)&u[base] = x;
}

extern "C" void kernel_launch(void* const* d_in, const int* in_sizes, int n_in,
                              void* d_out, int out_size, void* d_ws, size_t ws_size,
                              hipStream_t stream) {
  const float* q  = (const float*)d_in[0];
  const float* k  = (const float*)d_in[1];
  const float* v  = (const float*)d_in[2];
  const float* Wq = (const float*)d_in[3];
  const float* bq = (const float*)d_in[4];
  const float* Wk = (const float*)d_in[5];
  const float* bk = (const float*)d_in[6];
  const float* Wv = (const float*)d_in[7];
  const float* bv = (const float*)d_in[8];
  const float* Wo = (const float*)d_in[9];
  const float* bo = (const float*)d_in[10];

  float* attn = (float*)d_out;                                  // [16,2048,2048] fp32
  float* outf = attn + (long long)N_HEAD * N_TOK * N_TOK;       // [2048,1024] fp32

  char* ws = (char*)d_ws;
  bf16_t* WqT = (bf16_t*)ws; ws += (size_t)D_MODEL * D_MODEL * sizeof(bf16_t);
  bf16_t* WkT = (bf16_t*)ws; ws += (size_t)D_MODEL * D_MODEL * sizeof(bf16_t);
  bf16_t* WvT = (bf16_t*)ws; ws += (size_t)D_MODEL * D_MODEL * sizeof(bf16_t);
  bf16_t* WoT = (bf16_t*)ws; ws += (size_t)D_MODEL * D_MODEL * sizeof(bf16_t);
  bf16_t* Qb  = (bf16_t*)ws; ws += (size_t)N_TOK * D_MODEL * sizeof(bf16_t);
  bf16_t* Kb  = (bf16_t*)ws; ws += (size_t)N_TOK * D_MODEL * sizeof(bf16_t);
  bf16_t* Vb  = (bf16_t*)ws; ws += (size_t)N_TOK * D_MODEL * sizeof(bf16_t);
  bf16_t* Vt  = (bf16_t*)ws; ws += (size_t)D_MODEL * N_TOK * sizeof(bf16_t);   // [H*d, n_k]
  float*  outh = (float*)ws; ws += (size_t)N_HEAD * N_TOK * D_HEAD * sizeof(float); // [H,n_q,d]
  float*  mxA  = (float*)ws; ws += (size_t)N_HEAD * N_TOK * sizeof(float);
  float*  smA  = (float*)ws; ws += (size_t)N_HEAD * N_TOK * sizeof(float);

  const dim3 tb(32, 8);
  // 1) Weights -> bf16 transposed [out,in]
  transpose_to_bf16<float><<<dim3(D_MODEL/32, D_MODEL/32), tb, 0, stream>>>(Wq, WqT, D_MODEL, D_MODEL);
  transpose_to_bf16<float><<<dim3(D_MODEL/32, D_MODEL/32), tb, 0, stream>>>(Wk, WkT, D_MODEL, D_MODEL);
  transpose_to_bf16<float><<<dim3(D_MODEL/32, D_MODEL/32), tb, 0, stream>>>(Wv, WvT, D_MODEL, D_MODEL);
  transpose_to_bf16<float><<<dim3(D_MODEL/32, D_MODEL/32), tb, 0, stream>>>(Wo, WoT, D_MODEL, D_MODEL);

  // 2) QKV projections: [2048,1024]x[1024,1024] + bias -> bf16
  gemm_bf16_wmma<128,128,32,64,float,bf16_t><<<dim3(N_TOK/128, D_MODEL/128, 1), 256, 0, stream>>>(
      q, 0, D_MODEL, WqT, 0, D_MODEL, Qb, 0, D_MODEL, bq, D_MODEL, 1.0f);
  gemm_bf16_wmma<128,128,32,64,float,bf16_t><<<dim3(N_TOK/128, D_MODEL/128, 1), 256, 0, stream>>>(
      k, 0, D_MODEL, WkT, 0, D_MODEL, Kb, 0, D_MODEL, bk, D_MODEL, 1.0f);
  gemm_bf16_wmma<128,128,32,64,float,bf16_t><<<dim3(N_TOK/128, D_MODEL/128, 1), 256, 0, stream>>>(
      v, 0, D_MODEL, WvT, 0, D_MODEL, Vb, 0, D_MODEL, bv, D_MODEL, 1.0f);

  // 3) V transposed per head: Vt[h*64+d][k] (full [2048,1024] -> [1024,2048] transpose)
  transpose_to_bf16<bf16_t><<<dim3(D_MODEL/32, N_TOK/32), tb, 0, stream>>>(Vb, Vt, N_TOK, D_MODEL);

  // 4) Scores u[h,q,k] = Qh . Kh^T / 8 -> fp32 into d_out attn region (batched over heads)
  //    Both operands bf16 -> fully async-staged, double-buffered.
  gemm_bf16_wmma<128,128,32,64,bf16_t,float><<<dim3(N_TOK/128, N_TOK/128, N_HEAD), 256, 0, stream>>>(
      Qb, D_HEAD, D_MODEL, Kb, D_HEAD, D_MODEL,
      attn, (long long)N_TOK * N_TOK, N_TOK, nullptr, D_HEAD, 0.125f);

  // 5) Softmax over the QUERY axis (columns), in place in d_out
  softmax_stats<<<dim3(N_TOK/32, N_HEAD), tb, 0, stream>>>(attn, mxA, smA);
  softmax_apply<<<(unsigned)((long long)N_HEAD * N_TOK * N_TOK / 4 / 256), 256, 0, stream>>>(attn, mxA, smA);

  // 6) out_h[h,q,d] = attn[h,q,:] . Vh[:,d]   (A fp32 from d_out, converted on the fly)
  gemm_bf16_wmma<128,64,16,64,float,float><<<dim3(N_TOK/128, 1, N_HEAD), 256, 0, stream>>>(
      attn, (long long)N_TOK * N_TOK, N_TOK,
      Vt, (long long)D_HEAD * N_TOK, N_TOK,
      outh, (long long)N_TOK * D_HEAD, D_HEAD, nullptr, N_TOK, 1.0f);

  // 7) Raw C-order reshape [H,n_q,d] -> [n_q, H*d] is just linear indexing of outh.
  //    out = outh_view @ Wo + bo
  gemm_bf16_wmma<128,128,32,64,float,float><<<dim3(N_TOK/128, D_MODEL/128, 1), 256, 0, stream>>>(
      outh, 0, D_MODEL, WoT, 0, D_MODEL, outf, 0, D_MODEL, bo, D_MODEL, 1.0f);
}
